// SimpleHeteroGNN_52450140619022
// MI455X (gfx1250) — compile-verified, matrix operand order checked by reading, not run on hardware
//
#include <hip/hip_runtime.h>
#include <math.h>

#define N_CELL 200000
#define N_WELL 512
#define E_CC   1600000
#define E_CW   65536
#define CF     12
#define WF     8
#define HD     64
#define NEG_SLOPE 0.2f

typedef __attribute__((ext_vector_type(16))) _Float16 v16h;
typedef __attribute__((ext_vector_type(8)))  _Float16 v8h;
typedef __attribute__((ext_vector_type(8)))  float    v8f;

// float atomic max via sign-aware integer atomics
__device__ __forceinline__ void atomicMaxF(float* addr, float v) {
    if (v >= 0.f) atomicMax((int*)addr, __float_as_int(v));
    else          atomicMin((unsigned int*)addr, __float_as_uint(v));
}

// ---------------- small dense GEMM (emb layers, well-side proj, MLP) ----------------
__global__ void k_small_gemm(const float* __restrict__ X, const float* __restrict__ W,
                             const float* __restrict__ b, float* __restrict__ out,
                             int M, int K, int N, int relu) {
    int idx = blockIdx.x * blockDim.x + threadIdx.x;
    if (idx >= M * N) return;
    int m = idx / N, n = idx % N;
    float acc = b ? b[n] : 0.f;
    const float* xr = X + (size_t)m * K;
    for (int k = 0; k < K; ++k) acc += xr[k] * W[(size_t)k * N + n];
    if (relu) acc = fmaxf(acc, 0.f);
    out[idx] = acc;
}

// ---------------- weight pre-transpose + f16 convert: Wt[mat][n][k] = W[mat][k][n] ----
__global__ void k_wt_f16(const float* __restrict__ W, _Float16* __restrict__ Wt, int nmat) {
    int idx = blockIdx.x * blockDim.x + threadIdx.x;
    if (idx >= nmat * HD * HD) return;
    int mat = idx / (HD * HD);
    int r   = idx % (HD * HD);
    int n = r / HD, k = r % HD;
    Wt[idx] = (_Float16)W[(size_t)mat * HD * HD + (size_t)k * HD + n];
}

// ---------------- WMMA GEMM: out[M x 64] = A[M x 64] @ W[64 x 64] ----------------
// One wave per 16x16 output tile; K=64 -> two chained v_wmma_f32_16x16x32_f16.
// A is f32 row-major; Wt is f16, pre-transposed [N][K] so B-fragment loads are contiguous.
__global__ void k_wmma_gemm64(const float* __restrict__ A, const _Float16* __restrict__ Wt,
                              float* __restrict__ out, int M) {
    int wave = blockIdx.x * (blockDim.x >> 5) + (threadIdx.x >> 5);
    int lane = threadIdx.x & 31;
    int mt = wave >> 2;           // 4 N-tiles (64/16)
    int nt = wave & 3;
    if (mt * 16 >= M) return;     // wave-uniform: EXEC stays all-1s for live waves
    int m0 = mt * 16, n0 = nt * 16;
    int half = lane >> 4;         // lane-half selects K sub-block per ISA layout
    int l16  = lane & 15;
    int row  = m0 + l16;
    if (row >= M) row = M - 1;    // branch-free clamp (padded rows never stored)
    const float* arow = A + (size_t)row * HD;            // 256B-aligned row
    const _Float16* wcol = Wt + (size_t)(n0 + l16) * HD; // 128B-aligned column
    v8f acc = {};
    #pragma unroll
    for (int kk = 0; kk < HD; kk += 32) {
        int kb = kk + half * 8;
        // A 16x32 f16 fragment: elems 0..7 -> K=kb..kb+7 ; elems 8..15 -> K=kb+16..kb+23
        v8f xa = *(const v8f*)(arow + kb);        // 32B vector load
        v8f xb = *(const v8f*)(arow + kb + 16);   // 32B vector load
        v8h ha = __builtin_convertvector(xa, v8h);
        v8h hb = __builtin_convertvector(xb, v8h);
        v16h a = __builtin_shufflevector(ha, hb, 0,1,2,3,4,5,6,7,8,9,10,11,12,13,14,15);
        // B 32x16 f16 fragment: same K pattern along pre-transposed column
        v8h b0 = *(const v8h*)(wcol + kb);        // 16B load
        v8h b1 = *(const v8h*)(wcol + kb + 16);   // 16B load
        v16h b = __builtin_shufflevector(b0, b1, 0,1,2,3,4,5,6,7,8,9,10,11,12,13,14,15);
        acc = __builtin_amdgcn_wmma_f32_16x16x32_f16(false, a, false, b,
                                                     (short)0, acc, false, false);
    }
    // C/D layout: element r of lane -> row m0 + r + 8*(lane/16), col n0 + lane%16
    float* op = out + (size_t)(m0 + 8 * half) * HD + n0 + l16;
    if (m0 + 16 <= M) {           // wave-uniform fast path: straight-line stores
        #pragma unroll
        for (int r = 0; r < 8; ++r) op[(size_t)r * HD] = acc[r];
    } else {                      // tail tile: per-row guard
        #pragma unroll
        for (int r = 0; r < 8; ++r) {
            if (m0 + r + 8 * half < M) op[(size_t)r * HD] = acc[r];
        }
    }
}

// ---------------- fused attention logits: asO[i]=dot(H[i],attA), adO[i]=dot(H[i],attB) --
__global__ void k_row_dot2(const float* __restrict__ Hm, const float* __restrict__ attA,
                           const float* __restrict__ attB, float* __restrict__ asO,
                           float* __restrict__ adO, int M) {
    int i = blockIdx.x * blockDim.x + threadIdx.x;
    if (i >= M) return;
    const float4* r = (const float4*)(Hm + (size_t)i * HD);
    const float4* a = (const float4*)attA;
    const float4* b = (const float4*)attB;
    float sa = 0.f, sb = 0.f;
    #pragma unroll
    for (int j = 0; j < HD / 4; ++j) {
        float4 x = r[j], ya = a[j], yb = b[j];
        sa += x.x * ya.x + x.y * ya.y + x.z * ya.z + x.w * ya.w;
        sb += x.x * yb.x + x.y * yb.y + x.z * yb.z + x.w * yb.w;
    }
    asO[i] = sa;
    adO[i] = sb;
}

__global__ void k_row_dot(const float* __restrict__ Hm, const float* __restrict__ att,
                          float* __restrict__ out, int M) {
    int i = blockIdx.x * blockDim.x + threadIdx.x;
    if (i >= M) return;
    const float4* r = (const float4*)(Hm + (size_t)i * HD);
    const float4* a = (const float4*)att;
    float s = 0.f;
    #pragma unroll
    for (int j = 0; j < HD / 4; ++j) {
        float4 x = r[j], y = a[j];
        s += x.x * y.x + x.y * y.y + x.z * y.z + x.w * y.w;
    }
    out[i] = s;
}

// ---------------- init dst accumulator (=bias) and segment max/sum ----------------
__global__ void k_init_dst(float* __restrict__ hnext, const float* __restrict__ bias,
                           float* __restrict__ segmax, float* __restrict__ segsum, int M) {
    int idx = blockIdx.x * blockDim.x + threadIdx.x;
    if (idx >= M * HD) return;
    int h = idx & (HD - 1);
    hnext[idx] = bias[h];
    if (h == 0) { int node = idx >> 6; segmax[node] = -INFINITY; segsum[node] = 0.f; }
}

// ---------------- edge scalar precompute: k_l = We @ lin_edge_l @ att_e_l (ED==1) ----
__global__ void k_edge_scalars(const float* __restrict__ We, const float* __restrict__ be,
                               const float* __restrict__ lin_edge,
                               const float* __restrict__ att_e, float* __restrict__ kv) {
    if (threadIdx.x != 0 || blockIdx.x != 0) return;
    for (int l = 0; l < 3; ++l) {
        float k = 0.f, c = 0.f;
        for (int j = 0; j < HD; ++j) {
            float tj = 0.f, bj = 0.f;
            for (int h = 0; h < HD; ++h) {
                float le = lin_edge[(size_t)l * HD * HD + (size_t)h * HD + j];
                tj += We[h] * le;
                bj += be[h] * le;
            }
            k += tj * att_e[l * HD + j];
            c += bj * att_e[l * HD + j];
        }
        kv[l] = k; kv[3 + l] = c;
    }
}

// ---------------- pass A: alpha = lrelu(as[src]+ad[dst]+k*attr+c); segmax ----------
__global__ void k_edge_alpha(const int* __restrict__ src, const int* __restrict__ dst,
                             const float* __restrict__ attr, const float* __restrict__ kp,
                             const float* __restrict__ cp,
                             const float* __restrict__ as_, const float* __restrict__ ad_,
                             float* __restrict__ alpha, float* __restrict__ segmax, int E) {
    int e = blockIdx.x * blockDim.x + threadIdx.x;
    if (e >= E) return;
    int d = dst[e];
    float a = as_[src[e]] + ad_[d];
    if (attr) a += kp[0] * attr[e] + cp[0];
    a = a > 0.f ? a : NEG_SLOPE * a;
    alpha[e] = a;
    atomicMaxF(&segmax[d], a);
}

// ---------------- pass B: exponentiate, segment sum ----------------
__global__ void k_edge_exp(const int* __restrict__ dst, float* __restrict__ alpha,
                           const float* __restrict__ segmax, float* __restrict__ segsum, int E) {
    int e = blockIdx.x * blockDim.x + threadIdx.x;
    if (e >= E) return;
    int d = dst[e];
    float ex = __expf(alpha[e] - segmax[d]);
    alpha[e] = ex;
    atomicAdd(&segsum[d], ex);
}

// ---------------- pass C: hnext[dst] += hs[src] * p (16 lanes x float4 per edge) ----
__global__ void k_edge_scatter(const int* __restrict__ src, const int* __restrict__ dst,
                               const float* __restrict__ alpha, const float* __restrict__ segsum,
                               const float* __restrict__ hs, float* __restrict__ hnext, int E) {
    int idx = blockIdx.x * blockDim.x + threadIdx.x;
    if (idx >= E * 16) return;
    int e  = idx >> 4;
    int c0 = (idx & 15) * 4;
    int d  = dst[e];
    float p = alpha[e] / segsum[d];
    float4 v = *(const float4*)(hs + (size_t)src[e] * HD + c0);
    float* o = hnext + (size_t)d * HD + c0;
    atomicAdd(o + 0, v.x * p);
    atomicAdd(o + 1, v.y * p);
    atomicAdd(o + 2, v.z * p);
    atomicAdd(o + 3, v.w * p);
}

__global__ void k_relu(float* __restrict__ x, int n) {
    int i = blockIdx.x * blockDim.x + threadIdx.x;
    if (i < n) x[i] = fmaxf(x[i], 0.f);
}

extern "C" void kernel_launch(void* const* d_in, const int* in_sizes, int n_in,
                              void* d_out, int out_size, void* d_ws, size_t ws_size,
                              hipStream_t stream) {
    const float* cell_x        = (const float*)d_in[0];
    const float* well_x        = (const float*)d_in[1];
    const float* attr          = (const float*)d_in[2];
    const int*   cc            = (const int*)d_in[3];
    const int*   cw            = (const int*)d_in[4];
    const float* W_cell        = (const float*)d_in[5];
    const float* b_cell        = (const float*)d_in[6];
    const float* W_well        = (const float*)d_in[7];
    const float* b_well        = (const float*)d_in[8];
    const float* W_edge        = (const float*)d_in[9];
    const float* b_edge        = (const float*)d_in[10];
    const float* conv_lin      = (const float*)d_in[11];
    const float* conv_lin_edge = (const float*)d_in[12];
    const float* att_src       = (const float*)d_in[13];
    const float* att_dst       = (const float*)d_in[14];
    const float* att_edge      = (const float*)d_in[15];
    const float* conv_bias     = (const float*)d_in[16];
    const float* wc_lin        = (const float*)d_in[17];
    const float* wc_att_src    = (const float*)d_in[18];
    const float* wc_att_dst    = (const float*)d_in[19];
    const float* wc_bias       = (const float*)d_in[20];
    const float* mlp_W1        = (const float*)d_in[21];
    const float* mlp_b1        = (const float*)d_in[22];
    const float* mlp_W2        = (const float*)d_in[23];
    const float* mlp_b2        = (const float*)d_in[24];

    const int* src_cc = cc;
    const int* dst_cc = cc + E_CC;
    const int* src_cw = cw;
    const int* dst_cw = cw + E_CW;

    float* w = (float*)d_ws;
    size_t off = 0;
    float* hbuf  = w + off; off += (size_t)N_CELL * HD;   // current h_c / next-layer accumulator
    float* hsbuf = w + off; off += (size_t)N_CELL * HD;   // hs = h_c @ lin
    float* alpha = w + off; off += E_CC;
    float* as_   = w + off; off += N_CELL;
    float* ad_   = w + off; off += N_CELL;
    float* smax  = w + off; off += N_CELL;
    float* ssum  = w + off; off += N_CELL;
    float* hw    = w + off; off += (size_t)N_WELL * HD;
    float* hdw   = w + off; off += (size_t)N_WELL * HD;
    float* hwo   = w + off; off += (size_t)N_WELL * HD;
    float* z1    = w + off; off += (size_t)N_WELL * HD;
    float* adw   = w + off; off += N_WELL;
    float* smaxw = w + off; off += N_WELL;
    float* ssumw = w + off; off += N_WELL;
    float* kv    = w + off; off += 8;
    off = (off + 7) & ~(size_t)7;                          // 32B-align the f16 area
    _Float16* wt_conv = (_Float16*)(w + off); off += (size_t)3 * HD * HD / 2; // 3 mats, f16
    _Float16* wt_wc   = (_Float16*)(w + off); off += (size_t)HD * HD / 2;
    (void)ws_size; (void)in_sizes; (void)n_in; (void)out_size;

    const int T = 256;
    auto cdiv = [](long a, long b) { return (int)((a + b - 1) / b); };

    // one-time weight transpose+convert, embeddings, edge scalars
    k_wt_f16<<<cdiv(3L * HD * HD, T), T, 0, stream>>>(conv_lin, wt_conv, 3);
    k_wt_f16<<<cdiv(1L * HD * HD, T), T, 0, stream>>>(wc_lin, wt_wc, 1);
    k_small_gemm<<<cdiv((long)N_CELL * HD, T), T, 0, stream>>>(cell_x, W_cell, b_cell, hbuf, N_CELL, CF, HD, 0);
    k_small_gemm<<<cdiv((long)N_WELL * HD, T), T, 0, stream>>>(well_x, W_well, b_well, hw, N_WELL, WF, HD, 0);
    k_edge_scalars<<<1, 32, 0, stream>>>(W_edge, b_edge, conv_lin_edge, att_edge, kv);

    int mTiles = (N_CELL + 15) / 16;
    int gemmBlocks = cdiv((long)mTiles * 4, 8);   // 8 waves/block, 1 tile/wave

    // 3 GAT conv layers over cells
    for (int l = 0; l < 3; ++l) {
        k_wmma_gemm64<<<gemmBlocks, 256, 0, stream>>>(hbuf, wt_conv + (size_t)l * HD * HD, hsbuf, N_CELL);
        k_row_dot2<<<cdiv(N_CELL, T), T, 0, stream>>>(hsbuf, att_src + l * HD, att_dst + l * HD, as_, ad_, N_CELL);
        k_init_dst<<<cdiv((long)N_CELL * HD, T), T, 0, stream>>>(hbuf, conv_bias + l * HD, smax, ssum, N_CELL);
        k_edge_alpha<<<cdiv(E_CC, T), T, 0, stream>>>(src_cc, dst_cc, attr, kv + l, kv + 3 + l, as_, ad_, alpha, smax, E_CC);
        k_edge_exp<<<cdiv(E_CC, T), T, 0, stream>>>(dst_cc, alpha, smax, ssum, E_CC);
        k_edge_scatter<<<cdiv((long)E_CC * 16, T), T, 0, stream>>>(src_cc, dst_cc, alpha, ssum, hsbuf, hbuf, E_CC);
        k_relu<<<cdiv((long)N_CELL * HD, T), T, 0, stream>>>(hbuf, N_CELL * HD);
    }

    // cell -> well GAT layer (no edge features, no relu)
    k_wmma_gemm64<<<gemmBlocks, 256, 0, stream>>>(hbuf, wt_wc, hsbuf, N_CELL);
    k_small_gemm<<<cdiv((long)N_WELL * HD, T), T, 0, stream>>>(hw, wc_lin, nullptr, hdw, N_WELL, HD, HD, 0);
    k_row_dot<<<cdiv(N_CELL, T), T, 0, stream>>>(hsbuf, wc_att_src, as_, N_CELL);
    k_row_dot<<<cdiv(N_WELL, T), T, 0, stream>>>(hdw, wc_att_dst, adw, N_WELL);
    k_init_dst<<<cdiv((long)N_WELL * HD, T), T, 0, stream>>>(hwo, wc_bias, smaxw, ssumw, N_WELL);
    k_edge_alpha<<<cdiv(E_CW, T), T, 0, stream>>>(src_cw, dst_cw, nullptr, nullptr, nullptr, as_, adw, alpha, smaxw, E_CW);
    k_edge_exp<<<cdiv(E_CW, T), T, 0, stream>>>(dst_cw, alpha, smaxw, ssumw, E_CW);
    k_edge_scatter<<<cdiv((long)E_CW * 16, T), T, 0, stream>>>(src_cw, dst_cw, alpha, ssumw, hsbuf, hwo, E_CW);

    // output MLP
    k_small_gemm<<<cdiv((long)N_WELL * HD, T), T, 0, stream>>>(hwo, mlp_W1, mlp_b1, z1, N_WELL, HD, HD, 1);
    k_small_gemm<<<cdiv((long)N_WELL * 75, T), T, 0, stream>>>(z1, mlp_W2, mlp_b2, (float*)d_out, N_WELL, HD, 75, 0);
}
